// DemopackDecoder_91044716741301
// MI455X (gfx1250) — matrix operand order repaired and checked by reading, not compiled, not run on hardware
//
#include <hip/hip_runtime.h>
#include <hip/hip_bf16.h>
#include <stdint.h>

typedef __attribute__((ext_vector_type(16))) __bf16 v16bf;
typedef __attribute__((ext_vector_type(8)))  __bf16 v8bf;
typedef __attribute__((ext_vector_type(4)))  __bf16 v4bf;
typedef __attribute__((ext_vector_type(8)))  float  v8f;
typedef __attribute__((ext_vector_type(4)))  float  v4f;
typedef __attribute__((ext_vector_type(4)))  int    v4i;

#define BM 128
#define BN 128
#define BK 32
#define BKP 40   // padded LDS row stride in halves (80B): conflict-free frag reads
#define DIM 4096

#if __has_builtin(__builtin_amdgcn_global_load_async_to_lds_b128)
#define HAVE_ASYNC_LDS 1
typedef __attribute__((address_space(1))) v4i gv4i;   // global int4
typedef __attribute__((address_space(3))) v4i lv4i;   // LDS int4
__device__ __forceinline__ void async_copy16(const void* g, void* l) {
  // 16B global -> LDS, tracked by ASYNCcnt, no VGPR data path.
  __builtin_amdgcn_global_load_async_to_lds_b128(
      (gv4i*)(uintptr_t)g, (lv4i*)(uint32_t)(uintptr_t)l, 0, 0);
}
#else
#define HAVE_ASYNC_LDS 0
#endif

__device__ __forceinline__ void split_bf16(float v, __bf16& h, __bf16& l) {
  h = (__bf16)v;                 // RNE high part
  l = (__bf16)(v - (float)h);    // exact residual
}

// A-fragment (16x32 MxK): lane m=lane&15, half=lane>>4; k = {h*8..h*8+7, 16+h*8..}.
__device__ __forceinline__ v16bf frag_a(const __bf16* lds, int rowBase, int lane) {
  const int m = lane & 15, h = lane >> 4;
  const __bf16* p = lds + (rowBase + m) * BKP;
  v8bf c0 = *(const v8bf*)(p + h * 8);
  v8bf c1 = *(const v8bf*)(p + 16 + h * 8);
  v16bf r;
#pragma unroll
  for (int i = 0; i < 8; ++i) { r[i] = c0[i]; r[i + 8] = c1[i]; }
  return r;
}

// B-fragment (32x16 KxN) from LDS stored transposed [n][BKP]: lane n=lane&15,
// k = {h*8..h*8+7, 16+h*8..16+h*8+7}, h = lane>>4.
__device__ __forceinline__ v16bf frag_b(const __bf16* lds, int colBase, int lane) {
  const int n = lane & 15, h = lane >> 4;
  const __bf16* p = lds + (colBase + n) * BKP;
  v8bf c0 = *(const v8bf*)(p + h * 8);
  v8bf c1 = *(const v8bf*)(p + 16 + h * 8);
  v16bf r;
#pragma unroll
  for (int i = 0; i < 8; ++i) { r[i] = c0[i]; r[i + 8] = c1[i]; }
  return r;
}

// bf16x3 compensated product: hi*hi + hi*lo + lo*hi, fp32 accumulate.
__device__ __forceinline__ v8f wmma3(v16bf ah, v16bf al, v16bf bh, v16bf bl, v8f c) {
  c = __builtin_amdgcn_wmma_f32_16x16x32_bf16(false, ah, false, bh, (short)0, c, false, false);
  c = __builtin_amdgcn_wmma_f32_16x16x32_bf16(false, ah, false, bl, (short)0, c, false, false);
  c = __builtin_amdgcn_wmma_f32_16x16x32_bf16(false, al, false, bh, (short)0, c, false, false);
  return c;
}

// ---------------------------------------------------------------------------
// Stage 1: W[o][e] = scale[t] * sum_d codewords[idx[t,r]][d] * rot[t][d][e]
// Software-pipelined: prefetch next K-tile into registers during WMMA.
// ---------------------------------------------------------------------------
__global__ __launch_bounds__(256)
void decode_weight_kernel(const float* __restrict__ codewords,
                          const int*   __restrict__ indices,
                          const float* __restrict__ rotations,
                          const float* __restrict__ scales,
                          __bf16* __restrict__ Whi,
                          __bf16* __restrict__ Wlo) {
  __shared__ __bf16 Ahi[BM * BKP], Alo[BM * BKP], Bhi[BN * BKP], Blo[BN * BKP];
  __shared__ int idxS[BM];

  const int t   = blockIdx.z;
  const int rb  = blockIdx.y;
  const int cb  = blockIdx.x;
  const int tid = threadIdx.x;
  const int lane = tid & 31;
  const int wave = tid >> 5;     // 8 waves: 4(m) x 2(n)
  const int wm = wave >> 1;
  const int wn = wave & 1;

  if (tid < BM) idxS[tid] = indices[t * 1024 + rb * BM + tid];
  __syncthreads();

  const float scale = scales[t];
  const float* rotT = rotations + (size_t)t * DIM * DIM;

  v8f acc[2][4] = {};
  v4f aR[4], bR[4];

  // Prologue: fetch K-tile 0 into registers.
#pragma unroll
  for (int c = 0; c < 4; ++c) {
    int vi = c * 256 + tid;
    aR[c] = *(const v4f*)(codewords + (size_t)idxS[vi >> 3] * DIM + (vi & 7) * 4);
    bR[c] = *(const v4f*)(rotT + (size_t)(vi >> 5) * DIM + cb * BN + (vi & 31) * 4);
  }

  for (int k0 = 0; k0 < DIM; k0 += BK) {
    __syncthreads();               // WAR: everyone done reading previous tile
    // Store prefetched registers into LDS (split hi/lo).
#pragma unroll
    for (int c = 0; c < 4; ++c) {
      int vi = c * 256 + tid;
      { int r = vi >> 3, ko = (vi & 7) * 4;
        v4bf h, l;
#pragma unroll
        for (int j = 0; j < 4; ++j) { __bf16 hh, ll; split_bf16(aR[c][j], hh, ll); h[j] = hh; l[j] = ll; }
        *(v4bf*)(Ahi + r * BKP + ko) = h;
        *(v4bf*)(Alo + r * BKP + ko) = l; }
      { int k = vi >> 5, n = (vi & 31) * 4;
#pragma unroll
        for (int j = 0; j < 4; ++j) {
          __bf16 hh, ll; split_bf16(bR[c][j], hh, ll);
          Bhi[(n + j) * BKP + k] = hh;
          Blo[(n + j) * BKP + k] = ll; } }
    }
    // Prefetch next K-tile while this one is consumed by WMMAs.
    if (k0 + BK < DIM) {
#pragma unroll
      for (int c = 0; c < 4; ++c) {
        int vi = c * 256 + tid;
        aR[c] = *(const v4f*)(codewords + (size_t)idxS[vi >> 3] * DIM + k0 + BK + (vi & 7) * 4);
        bR[c] = *(const v4f*)(rotT + (size_t)(k0 + BK + (vi >> 5)) * DIM + cb * BN + (vi & 31) * 4);
      }
    }
    __syncthreads();

#pragma unroll
    for (int im = 0; im < 2; ++im) {
      v16bf ah = frag_a(Ahi, wm * 32 + im * 16, lane);
      v16bf al = frag_a(Alo, wm * 32 + im * 16, lane);
#pragma unroll
      for (int in = 0; in < 4; ++in) {
        v16bf bh = frag_b(Bhi, wn * 64 + in * 16, lane);
        v16bf bl = frag_b(Blo, wn * 64 + in * 16, lane);
        acc[im][in] = wmma3(ah, al, bh, bl, acc[im][in]);
      }
    }
  }

  // Epilogue: scale, split to bf16 hi/lo planes in workspace.
#pragma unroll
  for (int im = 0; im < 2; ++im) {
#pragma unroll
    for (int in = 0; in < 4; ++in) {
#pragma unroll
      for (int r = 0; r < 8; ++r) {
        int o = t * 1024 + rb * BM + wm * 32 + im * 16 + (lane >> 4) * 8 + r;
        int e = cb * BN + wn * 64 + in * 16 + (lane & 15);
        float v = acc[im][in][r] * scale;
        __bf16 hh, ll; split_bf16(v, hh, ll);
        size_t off = (size_t)o * DIM + e;
        Whi[off] = hh;
        Wlo[off] = ll;
      }
    }
  }
}

// ---------------------------------------------------------------------------
// Stage 2: out[m][n] = sum_k X[m][k] * W[n][k] + bias[n]
// A: register-pipelined with on-the-fly split. B: already bf16 -> async
// global->LDS double buffer (ASYNCcnt), no VGPR staging.
// ---------------------------------------------------------------------------
__global__ __launch_bounds__(256)
void linear_kernel(const float* __restrict__ X,
                   const __bf16* __restrict__ Whi,
                   const __bf16* __restrict__ Wlo,
                   const float*  __restrict__ bias,
                   float* __restrict__ out) {
  __shared__ __bf16 Ahi[BM * BKP], Alo[BM * BKP];
  __shared__ __bf16 Bh[2][BN * BKP], Bl[2][BN * BKP];

  const int nb  = blockIdx.x;
  const int mb  = blockIdx.y;
  const int tid = threadIdx.x;
  const int lane = tid & 31;
  const int wave = tid >> 5;
  const int wm = wave >> 1;
  const int wn = wave & 1;

  v8f acc[2][4] = {};
  v4f aR[4];

  // Prologue: A tile 0 into registers; B tile 0 into LDS buffer 0.
#pragma unroll
  for (int c = 0; c < 4; ++c) {
    int vi = c * 256 + tid;
    aR[c] = *(const v4f*)(X + (size_t)(mb * BM + (vi >> 3)) * DIM + (vi & 7) * 4);
  }
#if HAVE_ASYNC_LDS
#pragma unroll
  for (int c = 0; c < 2; ++c) {
    int vi = c * 256 + tid;
    int n = vi >> 2, ko = (vi & 3) * 8;
    size_t g = (size_t)(nb * BN + n) * DIM + ko;
    async_copy16(Whi + g, &Bh[0][n * BKP + ko]);
    async_copy16(Wlo + g, &Bl[0][n * BKP + ko]);
  }
#endif

  int buf = 0;
  for (int k0 = 0; k0 < DIM; k0 += BK) {
    __syncthreads();               // WAR on A plane and B[buf^1]
    // Commit prefetched A registers to LDS (split hi/lo).
#pragma unroll
    for (int c = 0; c < 4; ++c) {
      int vi = c * 256 + tid;
      int r = vi >> 3, ko = (vi & 7) * 4;
      v4bf h, l;
#pragma unroll
      for (int j = 0; j < 4; ++j) { __bf16 hh, ll; split_bf16(aR[c][j], hh, ll); h[j] = hh; l[j] = ll; }
      *(v4bf*)(Ahi + r * BKP + ko) = h;
      *(v4bf*)(Alo + r * BKP + ko) = l;
    }
#if HAVE_ASYNC_LDS
    if (k0 + BK < DIM) {
      // Kick off next B tile into the other buffer, prefetch next A tile,
      // then drain this wave's 4 older async ops (tile k0 complete).
#pragma unroll
      for (int c = 0; c < 2; ++c) {
        int vi = c * 256 + tid;
        int n = vi >> 2, ko = (vi & 3) * 8;
        size_t g = (size_t)(nb * BN + n) * DIM + k0 + BK + ko;
        async_copy16(Whi + g, &Bh[buf ^ 1][n * BKP + ko]);
        async_copy16(Wlo + g, &Bl[buf ^ 1][n * BKP + ko]);
      }
#pragma unroll
      for (int c = 0; c < 4; ++c) {
        int vi = c * 256 + tid;
        aR[c] = *(const v4f*)(X + (size_t)(mb * BM + (vi >> 3)) * DIM + k0 + BK + (vi & 7) * 4);
      }
      asm volatile("s_wait_asynccnt 0x4" ::: "memory");
    } else {
      asm volatile("s_wait_asynccnt 0x0" ::: "memory");
    }
#else
    // Fallback: synchronous vector copy of B tile.
#pragma unroll
    for (int c = 0; c < 2; ++c) {
      int vi = c * 256 + tid;
      int n = vi >> 2, ko = (vi & 3) * 8;
      size_t g = (size_t)(nb * BN + n) * DIM + k0 + ko;
      *(v8bf*)(&Bh[buf][n * BKP + ko]) = *(const v8bf*)(Whi + g);
      *(v8bf*)(&Bl[buf][n * BKP + ko]) = *(const v8bf*)(Wlo + g);
    }
    if (k0 + BK < DIM) {
#pragma unroll
      for (int c = 0; c < 4; ++c) {
        int vi = c * 256 + tid;
        aR[c] = *(const v4f*)(X + (size_t)(mb * BM + (vi >> 3)) * DIM + k0 + BK + (vi & 7) * 4);
      }
    }
#endif
    __syncthreads();

#pragma unroll
    for (int im = 0; im < 2; ++im) {
      v16bf ah = frag_a(Ahi, wm * 32 + im * 16, lane);
      v16bf al = frag_a(Alo, wm * 32 + im * 16, lane);
#pragma unroll
      for (int in = 0; in < 4; ++in) {
        v16bf bh = frag_b(&Bh[buf][0], wn * 64 + in * 16, lane);
        v16bf bl = frag_b(&Bl[buf][0], wn * 64 + in * 16, lane);
        acc[im][in] = wmma3(ah, al, bh, bl, acc[im][in]);
      }
    }
    buf ^= 1;
  }

#pragma unroll
  for (int im = 0; im < 2; ++im) {
#pragma unroll
    for (int in = 0; in < 4; ++in) {
#pragma unroll
      for (int r = 0; r < 8; ++r) {
        int m = mb * BM + wm * 32 + im * 16 + (lane >> 4) * 8 + r;
        int n = nb * BN + wn * 64 + in * 16 + (lane & 15);
        out[(size_t)m * DIM + n] = acc[im][in][r] + bias[n];
      }
    }
  }
}

extern "C" void kernel_launch(void* const* d_in, const int* in_sizes, int n_in,
                              void* d_out, int out_size, void* d_ws, size_t ws_size,
                              hipStream_t stream) {
  const float* x         = (const float*)d_in[0];
  const float* codewords = (const float*)d_in[1];
  const int*   indices   = (const int*)d_in[2];
  const float* rotations = (const float*)d_in[3];
  const float* scales    = (const float*)d_in[4];
  const float* bias      = (const float*)d_in[5];
  float* out = (float*)d_out;

  __bf16* Whi = (__bf16*)d_ws;
  __bf16* Wlo = (__bf16*)((char*)d_ws + (size_t)DIM * DIM * sizeof(__bf16));

  decode_weight_kernel<<<dim3(32, 8, 4), 256, 0, stream>>>(
      codewords, indices, rotations, scales, Whi, Wlo);
  linear_kernel<<<dim3(32, 64), 256, 0, stream>>>(
      x, Whi, Wlo, bias, out);
}
